// GatedDeltaNet_78675210928581
// MI455X (gfx1250) — matrix operand order repaired and checked by reading, not compiled, not run on hardware
//
#include <hip/hip_runtime.h>
#include <hip/hip_bf16.h>
#include <stdint.h>

#define HID      2048
#define NVH      32
#define NKH      16
#define DK       128
#define DV       128
#define KCONV    4
#define KEY_DIM  2048
#define VAL_DIM  4096
#define CONV_DIM 8192
#define BATCH    2
#define SEQ      2048
#define ROWS     (BATCH * SEQ)   // 4096

typedef __attribute__((ext_vector_type(16))) __bf16 bf16x16_t;
typedef __attribute__((ext_vector_type(8)))  float  f32x8_t;

// Packed f32x2 -> bf16x2 via a single v_cvt_pk_bf16_f32 (no struct repack).
__device__ __forceinline__ unsigned int pack_bf16x2(float lo, float hi) {
  unsigned int u;
  asm("v_cvt_pk_bf16_f32 %0, %1, %2" : "=v"(u) : "v"(lo), "v"(hi));
  return u;
}

// ---------------------------------------------------------------------------
// NT GEMM: Y[M,N] = X[M,K] * W[N,K]^T   (f32 in/out, bf16 WMMA compute)
// 128x128 tile per block, 256 threads = 8 waves, each wave owns 16 rows.
// Software-pipelined: double-buffered LDS, global prefetch held in raw
// float4 regs so the loadcnt wait lands after the WMMA block; all 8 B
// fragments preloaded so the 8 WMMAs issue back-to-back.
// ---------------------------------------------------------------------------
__global__ __launch_bounds__(256) void gemm_nt_bf16_wmma(
    const float* __restrict__ X, const float* __restrict__ W,
    float* __restrict__ Y, int M, int N, int K) {
  (void)M;
  __shared__ alignas(16) unsigned short As[2][128 * 32];
  __shared__ alignas(16) unsigned short Bs[2][128 * 32];
  const int tid  = threadIdx.x;
  const int wave = tid >> 5;
  const int lane = tid & 31;
  const int half = lane >> 4;   // 0: lanes 0-15, 1: lanes 16-31
  const int l16  = lane & 15;
  const int bn = blockIdx.x, bm = blockIdx.y;

  f32x8_t acc[8] = {};

  const float* Xb = X + (size_t)bm * 128 * K;
  const float* Wt = W + (size_t)bn * 128 * K;

  struct Frag { union { unsigned int u[8]; bf16x16_t v; }; };

  // Per-thread staging slots: 4 float4 each from X and W per 128x32 tile.
  float4 xr[4], wr[4];

  auto fetch = [&](int k0) {
#pragma unroll
    for (int i = 0; i < 4; ++i) {
      const int id  = tid + 256 * i;      // 0..1023 float4 slots
      const int row = id >> 3;            // 0..127
      const int kq  = (id & 7) * 4;       // 0,4,...,28
      xr[i] = *(const float4*)(Xb + (size_t)row * K + k0 + kq);
      wr[i] = *(const float4*)(Wt + (size_t)row * K + k0 + kq);
    }
  };
  auto store_lds = [&](int buf) {
#pragma unroll
    for (int i = 0; i < 4; ++i) {
      const int id  = tid + 256 * i;
      const int row = id >> 3;
      const int kq  = (id & 7) * 4;
      *(uint2*)(&As[buf][row * 32 + kq]) =
          make_uint2(pack_bf16x2(xr[i].x, xr[i].y), pack_bf16x2(xr[i].z, xr[i].w));
      *(uint2*)(&Bs[buf][row * 32 + kq]) =
          make_uint2(pack_bf16x2(wr[i].x, wr[i].y), pack_bf16x2(wr[i].z, wr[i].w));
    }
  };

  fetch(0);
  store_lds(0);
  __syncthreads();

  const int nk = K >> 5;
  for (int t = 0; t < nk; ++t) {
    const int cur = t & 1;
    const bool more = (t + 1) < nk;
    if (more) fetch((t + 1) << 5);   // global prefetch; wait deferred to pack

    // A fragment (16x32 bf16): lanes 0-15 hold K 0..15, lanes 16-31
    // hold K 16..31 of row M = wave*16 + l16.
    Frag fa;
    const int arow = wave * 16 + l16;
    *(uint4*)(&fa.u[0]) = *(const uint4*)(&As[cur][arow * 32 + half * 16]);
    *(uint4*)(&fa.u[4]) = *(const uint4*)(&As[cur][arow * 32 + half * 16 + 8]);
    // B fragments (32x16 bf16): lane n<16 holds K 0..15 of column N=n,
    // lane n+16 holds K 16..31.  Preload all 8 so WMMAs can pipeline.
    Frag fb[8];
#pragma unroll
    for (int n = 0; n < 8; ++n) {
      const int brow = n * 16 + l16;
      *(uint4*)(&fb[n].u[0]) = *(const uint4*)(&Bs[cur][brow * 32 + half * 16]);
      *(uint4*)(&fb[n].u[4]) = *(const uint4*)(&Bs[cur][brow * 32 + half * 16 + 8]);
    }
#pragma unroll
    for (int n = 0; n < 8; ++n) {
      acc[n] = __builtin_amdgcn_wmma_f32_16x16x32_bf16(
          false, fa.v, false, fb[n].v, (short)0, acc[n], false, false);
    }

    if (more) store_lds(cur ^ 1);   // pack + write next tile (other buffer)
    __syncthreads();
  }

  // C/D layout: VGPR r -> M = r + half*8, N = l16 (per 16x16 tile).
  const int rbase = bm * 128 + wave * 16 + half * 8;
#pragma unroll
  for (int n = 0; n < 8; ++n) {
    const int col = bn * 128 + n * 16 + l16;
#pragma unroll
    for (int r = 0; r < 8; ++r) {
      Y[(size_t)(rbase + r) * N + col] = acc[n][r];
    }
  }
}

// ---------------------------------------------------------------------------
// Causal depthwise conv (K=4) along S + SiLU.  x layout: [B*S, CONV_DIM].
// ---------------------------------------------------------------------------
__global__ __launch_bounds__(256) void conv_silu_kernel(
    const float* __restrict__ xin, const float* __restrict__ cw,
    float* __restrict__ out) {
  const int idx = blockIdx.x * 256 + threadIdx.x;
  const int c = idx & (CONV_DIM - 1);
  const int s = (idx >> 13) & (SEQ - 1);
  const int b = idx >> 24;
  const float* col = xin + (size_t)b * SEQ * CONV_DIM + c;
  float acc = 0.f;
#pragma unroll
  for (int j = 0; j < KCONV; ++j) {
    const int sp = s - (KCONV - 1) + j;
    if (sp >= 0) acc += cw[c * KCONV + j] * col[(size_t)sp * CONV_DIM];
  }
  out[idx] = acc / (1.f + __expf(-acc));   // silu
}

// ---------------------------------------------------------------------------
// beta = sigmoid(h·Wb), g = -exp(A_log)*softplus(h·Wa + dt_bias).
// One block per row; threads 0-31 -> Wb, 32-63 -> Wa.
// ---------------------------------------------------------------------------
__global__ __launch_bounds__(64) void small_proj_kernel(
    const float* __restrict__ hidden, const float* __restrict__ Wbp,
    const float* __restrict__ Wap, const float* __restrict__ dt_bias,
    const float* __restrict__ A_log, float* __restrict__ betab,
    float* __restrict__ gbuf) {
  __shared__ float hrow[HID];
  const int row = blockIdx.x;
  const float* h = hidden + (size_t)row * HID;
  for (int i = threadIdx.x; i < HID; i += 64) hrow[i] = h[i];
  __syncthreads();
  const int n = threadIdx.x & 31;
  const float* wr = ((threadIdx.x < 32) ? Wbp : Wap) + (size_t)n * HID;
  float acc = 0.f;
  for (int i = 0; i < HID; ++i) acc += hrow[i] * wr[i];
  if (threadIdx.x < 32) {
    betab[row * NVH + n] = 1.f / (1.f + __expf(-acc));
  } else {
    const float x = acc + dt_bias[n];
    const float sp = (x > 20.f) ? x : log1pf(__expf(x));
    gbuf[row * NVH + n] = -__expf(A_log[n]) * sp;
  }
}

// ---------------------------------------------------------------------------
// Delta-rule recurrence, one block per (b, value-head). 512 threads:
// lane v = tid&127 (DV column), j = tid>>7 (owns DK slice j*32..j*32+31).
// State S[k,v] lives in registers (32 f32/thread). Fused: q/k l2norm,
// exp(g) decay, delta update, readout, gated RMSNorm * silu(z).
// ---------------------------------------------------------------------------
__global__ __launch_bounds__(512) void delta_scan_kernel(
    const float* __restrict__ qkv, const float* __restrict__ betab,
    const float* __restrict__ gbuf, const float* __restrict__ z,
    const float* __restrict__ norm_w, float* __restrict__ core_n) {
  __shared__ float qs[DK], ks[DK], vs[DV], dels[DV];
  __shared__ float redq[DK], redk[DK];
  __shared__ float part[4][DV];
  __shared__ float sc[4];   // [0]=rsq_q*scale, [1]=rsq_k / rstd, [2]=exp(g), [3]=beta

  const int tid = threadIdx.x;
  const int v = tid & 127;
  const int j = tid >> 7;              // 0..3
  const int bh = blockIdx.x;
  const int b = bh >> 5;
  const int h = bh & 31;
  const int kh = h >> 1;               // rep = NVH/NKH = 2

  float st[32];
#pragma unroll
  for (int i = 0; i < 32; ++i) st[i] = 0.f;

  for (int s = 0; s < SEQ; ++s) {
    const size_t row = (size_t)b * SEQ + s;
    const float* base = qkv + row * CONV_DIM;
    if (tid < 128) {
      const float qv = base[kh * DK + tid];
      qs[tid] = qv; redq[tid] = qv * qv;
    } else if (tid < 256) {
      const int t = tid - 128;
      const float kv = base[KEY_DIM + kh * DK + t];
      ks[t] = kv; redk[t] = kv * kv;
    } else if (tid < 384) {
      const int t = tid - 256;
      vs[t] = base[2 * KEY_DIM + h * DV + t];
    } else if (tid == 384) {
      sc[2] = __expf(gbuf[row * NVH + h]);
    } else if (tid == 385) {
      sc[3] = betab[row * NVH + h];
    }
    __syncthreads();
    if (tid < 32) {                      // l2norm(q) * 1/sqrt(DK)
      float sum = redq[tid] + redq[tid + 32] + redq[tid + 64] + redq[tid + 96];
      for (int m = 16; m > 0; m >>= 1) sum += __shfl_xor(sum, m, 32);
      if (tid == 0) sc[0] = rsqrtf(sum + 1e-6f) * 0.08838834764831845f;
    } else if (tid < 64) {               // l2norm(k)
      const int t = tid - 32;
      float sum = redk[t] + redk[t + 32] + redk[t + 64] + redk[t + 96];
      for (int m = 16; m > 0; m >>= 1) sum += __shfl_xor(sum, m, 32);
      if (t == 0) sc[1] = rsqrtf(sum + 1e-6f);
    }
    __syncthreads();
    const float rq = sc[0], rk = sc[1], eg = sc[2], bet = sc[3];
    float kk[32];
    float p = 0.f;
#pragma unroll
    for (int i = 0; i < 32; ++i) {       // decay state, partial k·S
      kk[i] = ks[j * 32 + i] * rk;
      st[i] *= eg;
      p += st[i] * kk[i];
    }
    part[j][v] = p;
    __syncthreads();
    if (tid < 128) {
      const float kv = part[0][tid] + part[1][tid] + part[2][tid] + part[3][tid];
      dels[tid] = (vs[tid] - kv) * bet;
    }
    __syncthreads();
    const float d = dels[v];
    float p2 = 0.f;
#pragma unroll
    for (int i = 0; i < 32; ++i) {       // rank-1 update, partial q·S
      st[i] += kk[i] * d;
      p2 += st[i] * (qs[j * 32 + i] * rq);
    }
    part[j][v] = p2;
    __syncthreads();
    if (tid < 128) {
      const float o = part[0][tid] + part[1][tid] + part[2][tid] + part[3][tid];
      dels[tid] = o;                     // reuse as output staging
      redq[tid] = o * o;
    }
    __syncthreads();
    if (tid < 32) {                      // RMSNorm variance
      float sum = redq[tid] + redq[tid + 32] + redq[tid + 64] + redq[tid + 96];
      for (int m = 16; m > 0; m >>= 1) sum += __shfl_xor(sum, m, 32);
      if (tid == 0) sc[1] = rsqrtf(sum * (1.f / 128.f) + 1e-6f);
    }
    __syncthreads();
    if (tid < 128) {                     // gated RMSNorm * silu(z)
      const float zv = z[row * VAL_DIM + h * DV + tid];
      float o = dels[tid] * sc[1] * norm_w[tid];
      o *= zv / (1.f + __expf(-zv));
      core_n[row * VAL_DIM + h * DV + tid] = o;
    }
    // Safe rollover without a barrier: next phase-0 touches only
    // qs/ks/vs/redq/redk/sc[2..3]; dels/sc[1] are rewritten only after
    // subsequent barriers.
  }
}

// ---------------------------------------------------------------------------
extern "C" void kernel_launch(void* const* d_in, const int* in_sizes, int n_in,
                              void* d_out, int out_size, void* d_ws, size_t ws_size,
                              hipStream_t stream) {
  (void)in_sizes; (void)n_in; (void)out_size; (void)ws_size;
  const float* hidden  = (const float*)d_in[0];
  const float* Wqkv    = (const float*)d_in[1];
  const float* Wz      = (const float*)d_in[2];
  const float* Wbp     = (const float*)d_in[3];
  const float* Wap     = (const float*)d_in[4];
  const float* conv_w  = (const float*)d_in[5];
  const float* dt_bias = (const float*)d_in[6];
  const float* A_log   = (const float*)d_in[7];
  const float* norm_w  = (const float*)d_in[8];
  const float* Wout    = (const float*)d_in[9];
  float* out = (float*)d_out;

  // Workspace layout (regionA is reused after the conv consumes qkv_pre):
  //   regionA : ROWS*CONV_DIM f32  -> qkv_pre, then [z | core_n]
  //   qkv     : ROWS*CONV_DIM f32  -> conv+silu output
  //   beta, g : ROWS*NVH f32 each
  float* regionA = (float*)d_ws;
  float* qkv     = regionA + (size_t)ROWS * CONV_DIM;
  float* betab   = qkv + (size_t)ROWS * CONV_DIM;
  float* gbuf    = betab + (size_t)ROWS * NVH;
  float* qkv_pre = regionA;
  float* zbuf    = regionA;                              // after conv
  float* core_n  = regionA + (size_t)ROWS * VAL_DIM;     // after conv

  // 1) mixed_qkv = hidden · Wqkv^T
  gemm_nt_bf16_wmma<<<dim3(CONV_DIM / 128, ROWS / 128), 256, 0, stream>>>(
      hidden, Wqkv, qkv_pre, ROWS, CONV_DIM, HID);
  // 2) beta / g small projections (independent of qkv)
  small_proj_kernel<<<ROWS, 64, 0, stream>>>(
      hidden, Wbp, Wap, dt_bias, A_log, betab, gbuf);
  // 3) causal depthwise conv + silu (frees regionA)
  conv_silu_kernel<<<((size_t)ROWS * CONV_DIM) / 256, 256, 0, stream>>>(
      qkv_pre, conv_w, qkv);
  // 4) z = hidden · Wz^T  (into freed regionA)
  gemm_nt_bf16_wmma<<<dim3(VAL_DIM / 128, ROWS / 128), 256, 0, stream>>>(
      hidden, Wz, zbuf, ROWS, VAL_DIM, HID);
  // 5) delta-rule scan + gated RMSNorm, 64 chains
  delta_scan_kernel<<<BATCH * NVH, 512, 0, stream>>>(
      qkv, betab, gbuf, zbuf, norm_w, core_n);
  // 6) out = core_n · Wout^T
  gemm_nt_bf16_wmma<<<dim3(HID / 128, ROWS / 128), 256, 0, stream>>>(
      core_n, Wout, out, ROWS, HID, VAL_DIM);
}
